// self_attention_75831942578341
// MI455X (gfx1250) — matrix operand order repaired and checked by reading, not compiled, not run on hardware
//
#include <hip/hip_runtime.h>
#include <stdint.h>

// ---------------- types & helpers ----------------
typedef __attribute__((ext_vector_type(16))) __bf16 v16bf;
typedef __attribute__((ext_vector_type(8)))  float  v8f;
typedef __attribute__((ext_vector_type(4)))  int    v4i_t;
typedef __attribute__((ext_vector_type(2)))  int    v2i_t;

union FragBF { uint32_t u[8]; v16bf v; };
union Q2H { uint4 q; unsigned short s[8]; };

__device__ __forceinline__ unsigned short f2bf(float x) {
  union { float f; uint32_t u; } c; c.f = x;
  uint32_t r = ((c.u >> 16) & 1u) + 0x7fffu;
  return (unsigned short)((c.u + r) >> 16);
}
__device__ __forceinline__ float bf2f(unsigned short h) {
  union { uint32_t u; float f; } c; c.u = ((uint32_t)h) << 16;
  return c.f;
}

__device__ __forceinline__ v8f wmma_bf16(v16bf a, v16bf b, v8f c) {
  return __builtin_amdgcn_wmma_f32_16x16x32_bf16(false, a, false, b, (short)0, c, false, false);
}

// ---- async global->LDS copies (CDNA5 GLOBAL_LOAD_ASYNC_TO_LDS_*, ASYNCcnt) ----
#define AS1 __attribute__((address_space(1)))
#define AS3 __attribute__((address_space(3)))

#if __has_builtin(__builtin_amdgcn_global_load_async_to_lds_b128)
#define HAVE_ASYNC128 1
#endif
#if __has_builtin(__builtin_amdgcn_global_load_async_to_lds_b64)
#define HAVE_ASYNC64 1
#endif

__device__ __forceinline__ void copy_b128_g2l(const unsigned short* g, unsigned short* l) {
#ifdef HAVE_ASYNC128
  __builtin_amdgcn_global_load_async_to_lds_b128(
      (AS1 v4i_t*)(uintptr_t)g, (AS3 v4i_t*)(uint32_t)(uintptr_t)l, 0, 0);
#else
  *(uint4*)l = *(const uint4*)g;
#endif
}
__device__ __forceinline__ void copy_b64_g2l(const unsigned short* g, unsigned short* l) {
#ifdef HAVE_ASYNC64
  __builtin_amdgcn_global_load_async_to_lds_b64(
      (AS1 v2i_t*)(uintptr_t)g, (AS3 v2i_t*)(uint32_t)(uintptr_t)l, 0, 0);
#else
  *(uint2*)l = *(const uint2*)g;
#endif
}
__device__ __forceinline__ void async_wait() {
#if defined(HAVE_ASYNC128) || defined(HAVE_ASYNC64)
#if __has_builtin(__builtin_amdgcn_s_wait_asynccnt)
  __builtin_amdgcn_s_wait_asynccnt(0);
#else
  asm volatile("s_wait_asynccnt 0x0" ::: "memory");
#endif
#endif
}

// A fragment (16x32 bf16, row-major in LDS). lane&15 = row, lane>>4 picks K half.
__device__ __forceinline__ v16bf frag_a_lds(const unsigned short* base, int pitch, int lane, int k0) {
  FragBF f;
  const unsigned short* p = base + (lane & 15) * pitch + k0 + ((lane >> 4) << 4);
  #pragma unroll
  for (int g = 0; g < 8; ++g) {
    int k = g << 1;   // g*2
    f.u[g] = *(const uint32_t*)(p + k);
  }
  return f.v;
}

// B fragment from pre-packed weight: 8 contiguous dwords per lane.
__device__ __forceinline__ v16bf frag_b_pk(const uint32_t* p) {
  FragBF f;
  #pragma unroll
  for (int g = 0; g < 8; ++g) f.u[g] = p[g];
  return f.v;
}

// ---------------- weight packing ----------------
// dst dword index = ((ntile*ksteps + kstep)*32 + lane)*8 + g
// B layout (32x16): lane&15 = n, K = kstep*32 + 16*(lane>>4) + 2g + {0,1}
__global__ void k_pack_weight(const float* __restrict__ W, uint32_t* __restrict__ out,
                              int K, int N, int total) {
  int i = blockIdx.x * blockDim.x + threadIdx.x;
  if (i >= total) return;
  int g    = i & 7;
  int lane = (i >> 3) & 31;
  int t    = i >> 8;
  int ksteps = K >> 5;
  int kstep = t % ksteps;
  int ntile = t / ksteps;
  int n  = ntile * 16 + (lane & 15);
  int kb = kstep * 32 + ((lane >> 4) << 4) + (g << 1);
  uint32_t lo = f2bf(W[(size_t)kb * N + n]);
  uint32_t hi = f2bf(W[(size_t)(kb + 1) * N + n]);
  out[i] = lo | (hi << 16);
}

// ---------------- small elementwise kernels ----------------
__global__ void k_f2bf(const float* __restrict__ in, unsigned short* __restrict__ out, int n) {
  int i = blockIdx.x * blockDim.x + threadIdx.x;
  if (i < n) out[i] = f2bf(in[i]);
}

// hidden = relu(coors @ pos_w1 + pos_b1), K=3 done in VALU
__global__ void k_pos_hidden(const float* __restrict__ coors, const float* __restrict__ w1,
                             const float* __restrict__ b1, unsigned short* __restrict__ out, int total) {
  int i = blockIdx.x * blockDim.x + threadIdx.x;
  if (i >= total) return;
  int j = i & 511;
  size_t row = (size_t)(i >> 9);
  float c0 = coors[row * 3], c1 = coors[row * 3 + 1], c2 = coors[row * 3 + 2];
  float h = c0 * w1[j] + c1 * w1[512 + j] + c2 * w1[1024 + j] + b1[j];
  out[i] = f2bf(h > 0.f ? h : 0.f);
}

// ---------------- epilogue functors ----------------
struct EpiFeatPos {            // feat_pos = C + b2 + feat   -> bf16
  const float* b2; const float* feat; unsigned short* out;
  __device__ void operator()(size_t row, int col, float v) const {
    out[row * 512 + col] = f2bf(v + b2[col] + feat[row * 512 + col]);
  }
};
struct EpiEluMask {            // (elu(C)+1)*mask -> bf16 (Q and K)
  const int* mask; unsigned short* out;
  __device__ void operator()(size_t row, int col, float v) const {
    float q = (v > 0.f) ? (v + 1.f) : __expf(v);
    out[row * 512 + col] = f2bf(q * (float)mask[row]);
  }
};
struct EpiScaleV {             // C / L -> bf16
  unsigned short* out;
  __device__ void operator()(size_t row, int col, float v) const {
    out[row * 512 + col] = f2bf(v * (1.0f / 512.0f));
  }
};
struct EpiRelu1024 {           // relu(C) -> bf16, ld=1024
  unsigned short* out;
  __device__ void operator()(size_t row, int col, float v) const {
    out[row * 1024 + col] = f2bf(v > 0.f ? v : 0.f);
  }
};

// ---------------- generic 64x64-tile bf16 GEMM ----------------
// C(row,col) = sum_k A[row,k] * B[k,col]; A = A1 for k<ksplit else A2 (concat support).
// A staged through LDS via async global->LDS copies; B from packed, L2-resident weights.
template <class Epi>
__global__ __launch_bounds__(256) void gemm_bf16(
    const unsigned short* __restrict__ A1, const unsigned short* __restrict__ A2,
    int ksplit, int ldA, const uint32_t* __restrict__ Bp, int K, Epi epi) {
  __shared__ unsigned short sA[64 * 72];    // pitch 72: 16B-aligned rows, conflict-light
  const int tid = threadIdx.x, lane = tid & 31, wave = tid >> 5;
  const int rsub  = wave & 3;    // row subtile (x16)
  const int cpair = wave >> 2;   // 0..1 -> 32 cols
  const size_t rowbase = (size_t)blockIdx.x * 64;
  const int colbase = blockIdx.y * 64;
  const int ksteps = K >> 5;
  v8f acc0 = {0,0,0,0,0,0,0,0}, acc1 = {0,0,0,0,0,0,0,0};
  const int lr = tid >> 2;            // 0..63
  const int lc = (tid & 3) * 16;      // elems
  for (int kc = 0; kc < K; kc += 64) {
    const unsigned short* src = (kc < ksplit)
        ? (A1 + (rowbase + lr) * (size_t)ldA + kc + lc)
        : (A2 + (rowbase + lr) * (size_t)ldA + (kc - ksplit) + lc);
    unsigned short* dst = sA + lr * 72 + lc;
    copy_b128_g2l(src, dst);
    copy_b128_g2l(src + 8, dst + 8);
    async_wait();
    __syncthreads();
    #pragma unroll
    for (int kk = 0; kk < 2; ++kk) {
      v16bf a = frag_a_lds(sA + rsub * 16 * 72, 72, lane, kk * 32);
      int kstep = (kc >> 5) + kk;
      int nt0 = (colbase >> 4) + cpair * 2;
      const uint32_t* b0 = Bp + (((size_t)nt0 * ksteps + kstep) * 32 + lane) * 8;
      const uint32_t* b1 = Bp + ((((size_t)nt0 + 1) * ksteps + kstep) * 32 + lane) * 8;
      acc0 = wmma_bf16(a, frag_b_pk(b0), acc0);
      acc1 = wmma_bf16(a, frag_b_pk(b1), acc1);
    }
    __syncthreads();
  }
  const size_t row0 = rowbase + rsub * 16 + (lane >> 4) * 8;
  const int col0 = colbase + cpair * 32 + (lane & 15);
  #pragma unroll
  for (int r = 0; r < 8; ++r) {
    epi(row0 + r, col0, acc0[r]);
    epi(row0 + r, col0 + 16, acc1[r]);
  }
}

// ---------------- per-(b,h) KV = K^T V  (+ Ksum) ----------------
// K/V chunks are stored TRANSPOSED in LDS ([d][s], pitch 34) so both WMMA fragment
// patterns become contiguous dword loads, and Ksum reads are contiguous.
__global__ __launch_bounds__(256) void k_kv(const unsigned short* __restrict__ Kb,
                                            const unsigned short* __restrict__ Vb,
                                            float* __restrict__ KVout, float* __restrict__ Ksum) {
  __shared__ unsigned short sKt[64 * 34];   // [d][s]
  __shared__ unsigned short sVt[64 * 34];   // [v][s]
  __shared__ float sKs[256];
  const int bh = blockIdx.x, b = bh >> 3, h = bh & 7;
  const int tid = threadIdx.x, lane = tid & 31, wave = tid >> 5;
  const int msub = wave & 3, npair = wave >> 2;
  v8f acc0 = {0,0,0,0,0,0,0,0}, acc1 = {0,0,0,0,0,0,0,0};
  const int lr = tid >> 3;            // s 0..31
  const int lc = (tid & 7) * 8;       // d0
  const int hi = lane >> 4;
  const int zd = tid & 63, zq = tid >> 6;   // Ksum partial: dim zd, s-quarter zq
  float ksp = 0.f;
  for (int s0 = 0; s0 < 512; s0 += 32) {
    size_t g = ((size_t)(b * 512 + s0 + lr)) * 512 + (size_t)h * 64 + lc;
    Q2H kx, vx;
    kx.q = *(const uint4*)(Kb + g);
    vx.q = *(const uint4*)(Vb + g);
    #pragma unroll
    for (int i = 0; i < 8; ++i) {
      sKt[(lc + i) * 34 + lr] = kx.s[i];
      sVt[(lc + i) * 34 + lr] = vx.s[i];
    }
    __syncthreads();
    #pragma unroll
    for (int s = 0; s < 8; ++s) ksp += bf2f(sKt[zd * 34 + zq * 8 + s]);
    FragBF fa, fb0, fb1;
    int d  = (lane & 15) + msub * 16;
    int n0 = (lane & 15) + npair * 32;
    #pragma unroll
    for (int g8 = 0; g8 < 8; ++g8) {
      int sa = (hi << 4) + (g8 << 1);                          // A pattern
      int sb = (hi << 4) + (g8 << 1);                          // B pattern
      fa.u[g8]  = *(const uint32_t*)(sKt + d * 34 + sa);
      fb0.u[g8] = *(const uint32_t*)(sVt + n0 * 34 + sb);
      fb1.u[g8] = *(const uint32_t*)(sVt + (n0 + 16) * 34 + sb);
    }
    acc0 = wmma_bf16(fa.v, fb0.v, acc0);
    acc1 = wmma_bf16(fa.v, fb1.v, acc1);
    __syncthreads();
  }
  sKs[zd * 4 + zq] = ksp;
  int m0 = msub * 16 + hi * 8;
  int n0 = npair * 32 + (lane & 15);
  float* out = KVout + (size_t)bh * 4096;
  #pragma unroll
  for (int r = 0; r < 8; ++r) {
    out[(m0 + r) * 64 + n0]      = acc0[r];
    out[(m0 + r) * 64 + n0 + 16] = acc1[r];
  }
  __syncthreads();
  if (tid < 64)
    Ksum[(size_t)bh * 64 + tid] = sKs[tid * 4] + sKs[tid * 4 + 1] + sKs[tid * 4 + 2] + sKs[tid * 4 + 3];
}

// pack KV (f32, per-bh 64x64) into bf16 B-fragment layout: bh*2048 dwords
__global__ void k_kvpack(const float* __restrict__ KV, uint32_t* __restrict__ out, int total) {
  int i = blockIdx.x * blockDim.x + threadIdx.x;
  if (i >= total) return;
  int g = i & 7, lane = (i >> 3) & 31;
  int t = (i >> 8) & 7;            // ntile*2 + kstep
  int bh = i >> 11;
  int kstep = t & 1, ntile = t >> 1;
  int n = ntile * 16 + (lane & 15);
  int k = kstep * 32 + ((lane >> 4) << 4) + (g << 1);
  const float* src = KV + (size_t)bh * 4096;
  uint32_t lo = f2bf(src[k * 64 + n]);
  uint32_t hi = f2bf(src[(k + 1) * 64 + n]);
  out[i] = lo | (hi << 16);
}

// ---------------- attention: out = (Q @ KV) * L/(Q.Ksum+eps) ----------------
__global__ __launch_bounds__(256) void k_attn(const unsigned short* __restrict__ Qb,
                                              const uint32_t* __restrict__ KVp,
                                              const float* __restrict__ Ksum,
                                              unsigned short* __restrict__ attn) {
  __shared__ unsigned short sQ[64 * 72];
  __shared__ float sZp[256];
  __shared__ float zinv[64];
  const int bh = blockIdx.x, b = bh >> 3, h = bh & 7;
  const int l0 = blockIdx.y * 64;
  const int tid = threadIdx.x, lane = tid & 31, wave = tid >> 5;
  const int msub = wave & 3, npair = wave >> 2;
  {
    int lr = tid >> 2, lc = (tid & 3) * 16;
    size_t g = ((size_t)(b * 512 + l0 + lr)) * 512 + (size_t)h * 64 + lc;
    const unsigned short* src = Qb + g;
    unsigned short* dst = sQ + lr * 72 + lc;
    copy_b128_g2l(src, dst);
    copy_b128_g2l(src + 8, dst + 8);
    async_wait();
  }
  __syncthreads();
  {   // Z = Q . Ksum, parallel over 256 threads (row = tid&63, quarter = tid>>6)
    int zr = tid & 63, zq = tid >> 6;
    const float* kz = Ksum + (size_t)bh * 64 + zq * 16;
    float zp = 0.f;
    #pragma unroll
    for (int d = 0; d < 16; ++d) zp += bf2f(sQ[zr * 72 + zq * 16 + d]) * kz[d];
    sZp[zr * 4 + zq] = zp;
  }
  __syncthreads();
  if (tid < 64) {
    float z = sZp[tid * 4] + sZp[tid * 4 + 1] + sZp[tid * 4 + 2] + sZp[tid * 4 + 3];
    zinv[tid] = 512.0f / (z + 1e-6f);
  }
  __syncthreads();
  v8f acc0 = {0,0,0,0,0,0,0,0}, acc1 = {0,0,0,0,0,0,0,0};
  const uint32_t* Bp = KVp + (size_t)bh * 2048;
  #pragma unroll
  for (int kstep = 0; kstep < 2; ++kstep) {
    v16bf a = frag_a_lds(sQ + msub * 16 * 72, 72, lane, kstep * 32);
    int nt0 = npair * 2;
    const uint32_t* b0 = Bp + (((size_t)nt0 * 2 + kstep) * 32 + lane) * 8;
    const uint32_t* b1 = Bp + (((size_t)(nt0 + 1) * 2 + kstep) * 32 + lane) * 8;
    acc0 = wmma_bf16(a, frag_b_pk(b0), acc0);
    acc1 = wmma_bf16(a, frag_b_pk(b1), acc1);
  }
  int ml = msub * 16 + (lane >> 4) * 8;
  int n0 = npair * 32 + (lane & 15);
  #pragma unroll
  for (int r = 0; r < 8; ++r) {
    size_t row = (size_t)(b * 512 + l0 + ml + r);
    float zi = zinv[ml + r];
    attn[row * 512 + h * 64 + n0]      = f2bf(acc0[r] * zi);
    attn[row * 512 + h * 64 + n0 + 16] = f2bf(acc1[r] * zi);
  }
}

// ---------------- full-row (16x512) GEMM + LayerNorm [+ residual] ----------------
template <bool RESID>
__global__ __launch_bounds__(256) void gemm_rowln(
    const unsigned short* __restrict__ A, int ldA, const uint32_t* __restrict__ Bp, int K,
    const float* __restrict__ gam, const float* __restrict__ bet,
    const float* __restrict__ feat, float* __restrict__ outF, unsigned short* __restrict__ outB) {
  __shared__ unsigned short sA[16 * 68];
  __shared__ float sC[16 * 516];
  __shared__ float sP[16 * 16], sP2[16 * 16];
  __shared__ float sMu[16], sRs[16];
  const int tid = threadIdx.x, lane = tid & 31, wave = tid >> 5;
  const size_t rowbase = (size_t)blockIdx.x * 16;
  const int ksteps = K >> 5;
  v8f acc[4];
  v8f z8 = {0,0,0,0,0,0,0,0};
  #pragma unroll
  for (int j = 0; j < 4; ++j) acc[j] = z8;
  const int lr = tid >> 4, lc = (tid & 15) * 4;   // 16 rows x 64 elems
  for (int kc = 0; kc < K; kc += 64) {
    copy_b64_g2l(A + (rowbase + lr) * (size_t)ldA + kc + lc, sA + lr * 68 + lc);
    async_wait();
    __syncthreads();
    #pragma unroll
    for (int kk = 0; kk < 2; ++kk) {
      v16bf a = frag_a_lds(sA, 68, lane, kk * 32);
      int kstep = (kc >> 5) + kk;
      #pragma unroll
      for (int j = 0; j < 4; ++j) {
        int ntile = wave * 4 + j;
        const uint32_t* bp = Bp + (((size_t)ntile * ksteps + kstep) * 32 + lane) * 8;
        acc[j] = wmma_bf16(a, frag_b_pk(bp), acc[j]);
      }
    }
    __syncthreads();
  }
  {
    int m0 = (lane >> 4) * 8;
    #pragma unroll
    for (int j = 0; j < 4; ++j) {
      int col = wave * 64 + j * 16 + (lane & 15);
      #pragma unroll
      for (int r = 0; r < 8; ++r) sC[(m0 + r) * 516 + col] = acc[j][r];
    }
  }
  __syncthreads();
  {
    int row = tid >> 4, seg = tid & 15;
    float s = 0.f, s2 = 0.f;
    #pragma unroll
    for (int c = 0; c < 32; ++c) { float x = sC[row * 516 + seg * 32 + c]; s += x; s2 += x * x; }
    sP[row * 16 + seg] = s; sP2[row * 16 + seg] = s2;
  }
  __syncthreads();
  if (tid < 16) {
    float s = 0.f, s2 = 0.f;
    #pragma unroll
    for (int j = 0; j < 16; ++j) { s += sP[tid * 16 + j]; s2 += sP2[tid * 16 + j]; }
    float mu = s * (1.0f / 512.0f);
    float var = s2 * (1.0f / 512.0f) - mu * mu;
    sMu[tid] = mu; sRs[tid] = rsqrtf(var + 1e-5f);
  }
  __syncthreads();
  {
    int row = tid >> 4, seg = tid & 15;
    size_t grow = rowbase + row;
    float mu = sMu[row], rs = sRs[row];
    #pragma unroll
    for (int c = 0; c < 32; ++c) {
      int col = seg * 32 + c;
      float y = (sC[row * 516 + col] - mu) * rs * gam[col] + bet[col];
      if (RESID) outF[grow * 512 + col] = feat[grow * 512 + col] + y;
      else       outB[grow * 512 + col] = f2bf(y);
    }
  }
}

// ---------------- host launcher ----------------
extern "C" void kernel_launch(void* const* d_in, const int* in_sizes, int n_in,
                              void* d_out, int out_size, void* d_ws, size_t ws_size,
                              hipStream_t stream) {
  (void)in_sizes; (void)n_in; (void)out_size; (void)ws_size;
  const float* feat  = (const float*)d_in[0];
  const float* coors = (const float*)d_in[1];
  const int*   mask  = (const int*)d_in[2];
  const float* pw1 = (const float*)d_in[3];
  const float* pb1 = (const float*)d_in[4];
  const float* pw2 = (const float*)d_in[5];
  const float* pb2 = (const float*)d_in[6];
  const float* wq  = (const float*)d_in[7];
  const float* wk  = (const float*)d_in[8];
  const float* wv  = (const float*)d_in[9];
  const float* wm  = (const float*)d_in[10];
  const float* w1  = (const float*)d_in[11];
  const float* w2  = (const float*)d_in[12];
  const float* n1g = (const float*)d_in[13];
  const float* n1b = (const float*)d_in[14];
  const float* n2g = (const float*)d_in[15];
  const float* n2b = (const float*)d_in[16];
  float* out = (float*)d_out;

  char* ws = (char*)d_ws;
  const size_t MB = 1ull << 20;
  unsigned short* featbf  = (unsigned short*)(ws);              // 64 MiB, alive whole launch
  unsigned short* featpos = (unsigned short*)(ws + 64  * MB);   // then reused as msg
  unsigned short* hidden  = (unsigned short*)(ws + 128 * MB);   // then reused as attn out
  unsigned short* Qb      = (unsigned short*)(ws + 192 * MB);   // then h2 (with Kb region)
  unsigned short* Kb      = (unsigned short*)(ws + 256 * MB);
  unsigned short* Vb      = (unsigned short*)(ws + 320 * MB);
  float*     KVf  = (float*)   (ws + 384 * MB);                 // 16 MiB
  uint32_t*  KVp  = (uint32_t*)(ws + 400 * MB);                 // 8 MiB
  float*     Ksum = (float*)   (ws + 408 * MB);                 // 256 KiB
  uint32_t*  pw2p = (uint32_t*)(ws + 409 * MB);
  uint32_t*  wqp  = pw2p + 131072;
  uint32_t*  wkp  = wqp  + 131072;
  uint32_t*  wvp  = wkp  + 131072;
  uint32_t*  wmp  = wvp  + 131072;
  uint32_t*  w1p  = wmp  + 131072;    // 524288 dwords
  uint32_t*  w2p  = w1p  + 524288;    // 262144 dwords
  unsigned short* msg  = featpos;
  unsigned short* attn = hidden;
  unsigned short* h2   = Qb;

  const int NE = 128 * 512 * 512;     // 33,554,432
  dim3 blk(256);

  // pack weights into WMMA B-fragment order (bf16)
  k_pack_weight<<<dim3(131072/256), blk, 0, stream>>>(pw2, pw2p, 512, 512, 131072);
  k_pack_weight<<<dim3(131072/256), blk, 0, stream>>>(wq,  wqp,  512, 512, 131072);
  k_pack_weight<<<dim3(131072/256), blk, 0, stream>>>(wk,  wkp,  512, 512, 131072);
  k_pack_weight<<<dim3(131072/256), blk, 0, stream>>>(wv,  wvp,  512, 512, 131072);
  k_pack_weight<<<dim3(131072/256), blk, 0, stream>>>(wm,  wmp,  512, 512, 131072);
  k_pack_weight<<<dim3(524288/256), blk, 0, stream>>>(w1,  w1p, 1024, 1024, 524288);
  k_pack_weight<<<dim3(262144/256), blk, 0, stream>>>(w2,  w2p, 1024, 512, 262144);

  k_f2bf<<<dim3(NE/256), blk, 0, stream>>>(feat, featbf, NE);
  k_pos_hidden<<<dim3(NE/256), blk, 0, stream>>>(coors, pw1, pb1, hidden, NE);

  dim3 g512(1024, 8);
  // feat_pos = hidden @ pos_w2 + b2 + feat
  gemm_bf16<<<g512, blk, 0, stream>>>(hidden, hidden, 512, 512, pw2p, 512,
                                      EpiFeatPos{pb2, feat, featpos});
  // Q / K / V projections with fused elu+mask / scale epilogues
  gemm_bf16<<<g512, blk, 0, stream>>>(featpos, featpos, 512, 512, wqp, 512, EpiEluMask{mask, Qb});
  gemm_bf16<<<g512, blk, 0, stream>>>(featpos, featpos, 512, 512, wkp, 512, EpiEluMask{mask, Kb});
  gemm_bf16<<<g512, blk, 0, stream>>>(featpos, featpos, 512, 512, wvp, 512, EpiScaleV{Vb});

  // KV = K^T V per (b,h), plus Ksum
  k_kv<<<dim3(1024), blk, 0, stream>>>(Kb, Vb, KVf, Ksum);
  k_kvpack<<<dim3(2097152/256), blk, 0, stream>>>(KVf, KVp, 2097152);

  // out = Q @ KV * L/(Q.Ksum+eps)
  k_attn<<<dim3(1024, 8), blk, 0, stream>>>(Qb, KVp, Ksum, attn);

  // msg = LN1(attn @ wm)
  gemm_rowln<false><<<dim3(4096), blk, 0, stream>>>(attn, 512, wmp, 512, n1g, n1b,
                                                    nullptr, nullptr, msg);
  // h2 = relu([feat, msg] @ mlp_w1)
  gemm_bf16<<<dim3(1024, 16), blk, 0, stream>>>(featbf, msg, 512, 512, w1p, 1024, EpiRelu1024{h2});
  // out = feat + LN2(h2 @ mlp_w2)
  gemm_rowln<true><<<dim3(4096), blk, 0, stream>>>(h2, 1024, w2p, 1024, n2g, n2b,
                                                   feat, out, nullptr);
}
